// CommunityGCN_90108413870706
// MI455X (gfx1250) — matrix operand order repaired and checked by reading, not compiled, not run on hardware
//
#include <hip/hip_runtime.h>
#include <hip/hip_bf16.h>

// ---------------------------------------------------------------------------
// CommunityGCN forward on gfx1250 (CDNA5, wave32, WMMA bf16)
//   x[N,128], edge_index[2,E], community[N]
//   h  = relu(concat(x, commMean(x)) @ W_in + b_in)            N x 128
//   h  = relu(gcn(h, W1, b1))                                  N x 128
//   out=        gcn(h, W2, b2)                                 N x 40
// gcn(h,W,b): hw = h@W; agg = scatter_add(hw[src]*dis[src]*dis[dst] -> dst);
//             agg + hw/deg + b
// ---------------------------------------------------------------------------

#define IN_C   128
#define HID    128
#define OUT_C  40
#define OUT_CP 48      // classifier cols padded to 3 WMMA tiles
#define NCOMM  1000

typedef __bf16 bf16x8  __attribute__((ext_vector_type(8)));
typedef __bf16 bf16x16 __attribute__((ext_vector_type(16)));
typedef float  f32x8   __attribute__((ext_vector_type(8)));

union BF16Frag { bf16x16 v; bf16x8 h[2]; };

// ------------------------------- elementwise -------------------------------

__global__ void zero_f32(float* __restrict__ p, int n) {
    int i = blockIdx.x * blockDim.x + threadIdx.x;
    if (i < n) p[i] = 0.0f;
}

__global__ void count_atomic(const int* __restrict__ idx, float* __restrict__ cnt, int n) {
    int i = blockIdx.x * blockDim.x + threadIdx.x;
    if (i < n) unsafeAtomicAdd(&cnt[idx[i]], 1.0f);
}

__global__ void comm_accum(const float* __restrict__ x, const int* __restrict__ com,
                           float* __restrict__ csum, int total) {
    int i = blockIdx.x * blockDim.x + threadIdx.x;
    if (i >= total) return;
    int n = i >> 7, j = i & 127;
    unsafeAtomicAdd(&csum[com[n] * IN_C + j], x[i]);
}

// A0 = bf16(concat(x, cmean[community]))  -> [N, 256] row-major
__global__ void build_a0(const float* __restrict__ x, const int* __restrict__ com,
                         const float* __restrict__ csum, const float* __restrict__ ccnt,
                         __bf16* __restrict__ abf, int total) {
    int i = blockIdx.x * blockDim.x + threadIdx.x;
    if (i >= total) return;
    int n = i >> 8, j = i & 255;
    float v;
    if (j < IN_C) {
        v = x[n * IN_C + j];
    } else {
        int c = com[n];
        float cnt = fmaxf(ccnt[c], 1.0f);
        v = csum[c * IN_C + (j - IN_C)] / cnt;
    }
    abf[i] = (__bf16)v;
}

// deg_io holds in-degree(dst); produce dis = rsqrt(deg+1), deg_io = 1/(deg+1)
__global__ void deg_finalize(float* __restrict__ deg_io, float* __restrict__ dis, int n) {
    int i = blockIdx.x * blockDim.x + threadIdx.x;
    if (i >= n) return;
    float d = deg_io[i] + 1.0f;
    dis[i] = rsqrtf(d);
    deg_io[i] = 1.0f / d;
}

__global__ void cvt_f32_bf16(const float* __restrict__ s, __bf16* __restrict__ d, int n) {
    int i = blockIdx.x * blockDim.x + threadIdx.x;
    if (i < n) d[i] = (__bf16)s[i];
}

// h2 = relu(agg + hw*invdeg + b1) stored directly as bf16 A-matrix
__global__ void combine_relu_bf16(const float* __restrict__ agg, const float* __restrict__ hw,
                                  const float* __restrict__ invdeg, const float* __restrict__ b,
                                  __bf16* __restrict__ abf, int total) {
    int i = blockIdx.x * blockDim.x + threadIdx.x;
    if (i >= total) return;
    int n = i >> 7, j = i & 127;
    float v = agg[i] + hw[i] * invdeg[n] + b[j];
    abf[i] = (__bf16)fmaxf(v, 0.0f);
}

// out = agg2 + hw2*invdeg + b2   (hw2 row stride OUT_CP, agg2/out stride OUT_C)
__global__ void final_out(const float* __restrict__ agg, const float* __restrict__ hw,
                          const float* __restrict__ invdeg, const float* __restrict__ b,
                          float* __restrict__ out, int total) {
    int i = blockIdx.x * blockDim.x + threadIdx.x;
    if (i >= total) return;
    int n = i / OUT_C;
    int c = i - n * OUT_C;
    out[i] = agg[i] + hw[n * OUT_CP + c] * invdeg[n] + b[c];
}

// ------------------------ weight packing (B fragments) ---------------------
// 16-bit B 32x16 fragment as packed here (k contiguous per lane half, same order as A), wave32:
//   lanes 0-15 : col = nt*16+lane,    K = kt*32 + j      (j=0..15)
//   lanes 16-31: col = nt*16+lane-16, K = kt*32 + 16 + j
// Bp[((kt*NTtot + nt)*32 + lane)*16 + j]
__global__ void pack_b(const float* __restrict__ W, __bf16* __restrict__ Bp,
                       int K, int Hreal, int Hpad) {
    int t = blockIdx.x * blockDim.x + threadIdx.x;
    int NTtot = Hpad >> 4;
    int total = (K >> 5) * NTtot * 32;
    if (t >= total) return;
    int lane = t & 31;
    int rest = t >> 5;
    int nt = rest % NTtot;
    int kt = rest / NTtot;
    int col = nt * 16 + (lane & 15);
    int kb = kt * 32 + ((lane >> 4) << 4);
    #pragma unroll
    for (int j = 0; j < 16; ++j) {
        float v = (col < Hreal) ? W[(kb + j) * Hreal + col] : 0.0f;
        Bp[(size_t)t * 16 + j] = (__bf16)v;
    }
}

// ------------------------------ WMMA GEMM ----------------------------------
// C[M,Ncols] = relu?(A[M,K]@B + bias). A bf16 row-major, B pre-packed.
// 1 wave -> 16 rows x (16*NT) cols. 4 waves/block over consecutive M tiles.
template<int NT>
__global__ __launch_bounds__(128)
void gemm_bf16_wmma(const __bf16* __restrict__ A, const __bf16* __restrict__ Bp,
                    const float* __restrict__ bias, float* __restrict__ C,
                    int M, int K, int Ncols, int relu) {
    const int lane = threadIdx.x & 31;
    const int wave = threadIdx.x >> 5;
    const int mt = blockIdx.x * 4 + wave;
    if (mt * 16 >= M) return;
    const int NTtot = Ncols >> 4;
    const int nt0 = blockIdx.y * NT;

    // A fragment per-lane base: lanes 0-15 row m, K 0..15; lanes 16-31 K 16..31
    int arow = mt * 16 + (lane & 15);
    if (arow >= M) arow = M - 1;
    const __bf16* Arow = A + (size_t)arow * K + ((lane >> 4) << 4);

    f32x8 acc[NT] = {};
    const int kTiles = K >> 5;
    for (int kt = 0; kt < kTiles; ++kt) {
        BF16Frag a;
        a.h[0] = *(const bf16x8*)(Arow + kt * 32);
        a.h[1] = *(const bf16x8*)(Arow + kt * 32 + 8);
        #pragma unroll
        for (int t = 0; t < NT; ++t) {
            const __bf16* bp = Bp + ((size_t)(kt * NTtot + nt0 + t) * 32 + lane) * 16;
            BF16Frag b;
            b.h[0] = *(const bf16x8*)(bp);
            b.h[1] = *(const bf16x8*)(bp + 8);
            acc[t] = __builtin_amdgcn_wmma_f32_16x16x32_bf16(
                false, a.v, false, b.v, (short)0, acc[t], false, false);
        }
    }

    // D layout: VGPR r, lane L -> row m0 + r + (L<16?0:8), col = n0 + (L&15)
    const int mrow = mt * 16 + ((lane >> 4) << 3);
    #pragma unroll
    for (int t = 0; t < NT; ++t) {
        int col = (nt0 + t) * 16 + (lane & 15);
        float bv = bias ? bias[col] : 0.0f;
        #pragma unroll
        for (int r = 0; r < 8; ++r) {
            int rr = mrow + r;
            if (rr < M) {
                float v = acc[t][r] + bv;
                if (relu) v = fmaxf(v, 0.0f);
                C[(size_t)rr * Ncols + col] = v;
            }
        }
    }
}

// --------------------------- edge scatter-add ------------------------------
// unit u = (edge e, 4-feature group g):
//   agg[dst[e], 4g..4g+3] += hw[src[e], 4g..4g+3] * dis[src]*dis[dst]
__global__ void edge_scatter(const int* __restrict__ src, const int* __restrict__ dst,
                             const float* __restrict__ dis, const float* __restrict__ hw,
                             float* __restrict__ agg, int RS, int AS, int F4,
                             unsigned units) {
    unsigned u = blockIdx.x * blockDim.x + threadIdx.x;
    if (u >= units) return;
    unsigned e = u / (unsigned)F4;
    unsigned g = u - e * (unsigned)F4;
    int s = src[e], d = dst[e];
    float norm = dis[s] * dis[d];
    const float4 v = *(const float4*)(hw + (size_t)s * RS + 4u * g);
    float* ap = agg + (size_t)d * AS + 4u * g;
    unsafeAtomicAdd(ap + 0, v.x * norm);
    unsafeAtomicAdd(ap + 1, v.y * norm);
    unsafeAtomicAdd(ap + 2, v.z * norm);
    unsafeAtomicAdd(ap + 3, v.w * norm);
}

// ------------------------------- launcher ----------------------------------

static inline unsigned nblk(long long n, int b) { return (unsigned)((n + b - 1) / b); }

extern "C" void kernel_launch(void* const* d_in, const int* in_sizes, int n_in,
                              void* d_out, int out_size, void* d_ws, size_t ws_size,
                              hipStream_t stream) {
    const float* x   = (const float*)d_in[0];
    const int*   edg = (const int*)d_in[1];
    const int*   com = (const int*)d_in[2];
    const float* Win = (const float*)d_in[3];
    const float* bin = (const float*)d_in[4];
    const float* W1  = (const float*)d_in[5];
    const float* b1  = (const float*)d_in[6];
    const float* W2  = (const float*)d_in[7];
    const float* b2  = (const float*)d_in[8];
    const int N = in_sizes[0] / IN_C;
    const int E = in_sizes[1] / 2;
    const int* src = edg;
    const int* dst = edg + E;

    // workspace carve-out (256B aligned)
    char* base = (char*)d_ws;
    size_t off = 0;
    auto carve = [&](size_t bytes) -> char* {
        char* p = base + off;
        off = (off + bytes + 255) & ~(size_t)255;
        return p;
    };
    float*  csum   = (float*)carve((size_t)NCOMM * IN_C * 4);
    float*  ccnt   = (float*)carve((size_t)NCOMM * 4);
    float*  dis    = (float*)carve((size_t)N * 4);
    float*  invdeg = (float*)carve((size_t)N * 4);           // also deg accumulator
    __bf16* WinP   = (__bf16*)carve((size_t)2 * IN_C * HID * 2);
    __bf16* W1P    = (__bf16*)carve((size_t)HID * HID * 2);
    __bf16* W2P    = (__bf16*)carve((size_t)HID * OUT_CP * 2);
    __bf16* ABF    = (__bf16*)carve((size_t)N * 2 * IN_C * 2); // A0 / h1 / h2 (bf16)
    float*  HBUF   = (float*)carve((size_t)N * HID * 4);       // h1 -> hw1 -> hw2
    float*  AGG    = (float*)carve((size_t)N * HID * 4);       // agg1 -> agg2

    const int TB = 256;

    // --- community mean + input assembly ---
    zero_f32<<<nblk(NCOMM * IN_C, TB), TB, 0, stream>>>(csum, NCOMM * IN_C);
    zero_f32<<<nblk(NCOMM, TB), TB, 0, stream>>>(ccnt, NCOMM);
    count_atomic<<<nblk(N, TB), TB, 0, stream>>>(com, ccnt, N);
    comm_accum<<<nblk((long long)N * IN_C, TB), TB, 0, stream>>>(x, com, csum, N * IN_C);
    build_a0<<<nblk((long long)N * 2 * IN_C, TB), TB, 0, stream>>>(x, com, csum, ccnt, ABF, N * 2 * IN_C);

    // --- pack weights into WMMA B-fragment layout ---
    pack_b<<<nblk((2 * IN_C / 32) * (HID / 16) * 32, TB), TB, 0, stream>>>(Win, WinP, 2 * IN_C, HID, HID);
    pack_b<<<nblk((HID / 32) * (HID / 16) * 32, TB), TB, 0, stream>>>(W1, W1P, HID, HID, HID);
    pack_b<<<nblk((HID / 32) * (OUT_CP / 16) * 32, TB), TB, 0, stream>>>(W2, W2P, HID, OUT_C, OUT_CP);

    // --- degrees ---
    zero_f32<<<nblk(N, TB), TB, 0, stream>>>(invdeg, N);
    count_atomic<<<nblk(E, TB), TB, 0, stream>>>(dst, invdeg, E);
    deg_finalize<<<nblk(N, TB), TB, 0, stream>>>(invdeg, dis, N);

    const int Mt = (N + 15) / 16;
    dim3 gemmBlk(128);
    dim3 gH((Mt + 3) / 4, HID / (16 * 4));   // NT=4 -> 64 cols per wave
    dim3 gC((Mt + 3) / 4, OUT_CP / (16 * 3)); // NT=3 -> 48 cols per wave

    // --- layer 0: h1 = relu(A0 @ W_in + b_in) ---
    gemm_bf16_wmma<4><<<gH, gemmBlk, 0, stream>>>(ABF, WinP, bin, HBUF, N, 2 * IN_C, HID, 1);
    cvt_f32_bf16<<<nblk((long long)N * HID, TB), TB, 0, stream>>>(HBUF, ABF, N * HID);

    // --- conv1: hw1 = h1 @ W1 ; scatter ; h2 = relu(agg + hw1/deg + b1) ---
    gemm_bf16_wmma<4><<<gH, gemmBlk, 0, stream>>>(ABF, W1P, nullptr, HBUF, N, HID, HID, 0);
    zero_f32<<<nblk((long long)N * HID, TB), TB, 0, stream>>>(AGG, N * HID);
    edge_scatter<<<nblk((long long)E * (HID / 4), TB), TB, 0, stream>>>(
        src, dst, dis, HBUF, AGG, HID, HID, HID / 4, (unsigned)E * (HID / 4));
    combine_relu_bf16<<<nblk((long long)N * HID, TB), TB, 0, stream>>>(AGG, HBUF, invdeg, b1, ABF, N * HID);

    // --- conv2: hw2 = h2 @ W2 (padded 48) ; scatter over 40 cols ; final ---
    gemm_bf16_wmma<3><<<gC, gemmBlk, 0, stream>>>(ABF, W2P, nullptr, HBUF, N, HID, OUT_CP, 0);
    zero_f32<<<nblk((long long)N * OUT_C, TB), TB, 0, stream>>>(AGG, N * OUT_C);
    edge_scatter<<<nblk((long long)E * (OUT_C / 4), TB), TB, 0, stream>>>(
        src, dst, dis, HBUF, AGG, OUT_CP, OUT_C, OUT_C / 4, (unsigned)E * (OUT_C / 4));
    final_out<<<nblk((long long)N * OUT_C, TB), TB, 0, stream>>>(AGG, HBUF, invdeg, b2, (float*)d_out, N * OUT_C);
}
